// Attention_79147657330728
// MI455X (gfx1250) — matrix operand order repaired and checked by reading, not compiled, not run on hardware
//
#include <hip/hip_runtime.h>
#include <hip/hip_bf16.h>

typedef __attribute__((ext_vector_type(16))) _Float16 v16h;
typedef __attribute__((ext_vector_type(8)))  _Float16 v8h;
typedef __attribute__((ext_vector_type(8)))  float    v8f;

// ---------------------------------------------------------------------------
// WMMA helpers for gfx1250: D = A(16x32 f16) * B(32x16 f16) + C(16x16 f32)
// A layout: lane t -> M = t&15 ; elements e=0..15 -> K = (t>>4)*16 + e
// B layout: lane t -> N = t&15 ; elements e=0..15 -> K = (t>>4)*16 + e
// C layout: lane t -> N = t&15 ; VGPR r -> M = r + 8*(t>>4)
// ---------------------------------------------------------------------------
__device__ __forceinline__ v8f wmma_f16(v16h a, v16h b, v8f c) {
    return __builtin_amdgcn_wmma_f32_16x16x32_f16(
        false, a, false, b, (short)0, c, false, false);
}

__device__ __forceinline__ v16h make_a16(v8h lo, v8h hi) {
    v16h a;
#pragma unroll
    for (int i = 0; i < 8; i++) { a[i] = lo[i]; a[i + 8] = hi[i]; }
    return a;
}

// Load A fragment from row-major f16 [16 rows x >=32 cols], base at (row0,k0).
__device__ __forceinline__ v16h load_a_tile(const _Float16* __restrict__ base,
                                            int ld, int lane) {
    int m  = lane & 15;
    int hi = lane >> 4;
    const _Float16* p = base + (size_t)m * ld + hi * 16;
    v8h lo = *(const v8h*)p;          // K = hi*16 .. hi*16+7      (16B)
    v8h hh = *(const v8h*)(p + 8);    // K = hi*16+8 .. hi*16+15   (16B)
    return make_a16(lo, hh);
}

// Load B fragment where memory is laid out [N rows][K cols] row-major
// (torch weight convention W[o][c], or K-matrix K[key][c]).
__device__ __forceinline__ v16h load_b_tile(const _Float16* __restrict__ base,
                                            int ld, int lane) {
    int n  = lane & 15;
    int hi = lane >> 4;
    return *(const v16h*)(base + (size_t)n * ld + hi * 16);   // 32B contiguous
}

// ---------------------------------------------------------------------------
// Async global->LDS copy: CDNA5 GLOBAL_LOAD_ASYNC_TO_LDS_B128 (ASYNCcnt)
// via inline asm (builtin signatures differ across toolchains). 16 B/lane.
//   vdst  = VGPR with 32-bit LDS byte address (per lane)
//   vaddr = VGPR pair with 64-bit global address, SADDR = off (null)
// ---------------------------------------------------------------------------
__device__ __forceinline__ void cp_g2l_16(const _Float16* g, _Float16* l) {
    unsigned int laddr =
        (unsigned int)(size_t)(__attribute__((address_space(3))) _Float16*)l;
    unsigned long long gaddr = (unsigned long long)(size_t)g;
    asm volatile("global_load_async_to_lds_b128 %0, %1, off"
                 :: "v"(laddr), "v"(gaddr) : "memory");
}

__device__ __forceinline__ void cp_g2l_wait() {
#if __has_builtin(__builtin_amdgcn_s_wait_asynccnt)
    __builtin_amdgcn_s_wait_asynccnt(0);
#else
    asm volatile("s_wait_asynccnt 0" ::: "memory");
#endif
}

// ---------------------------------------------------------------------------
// f32 -> f16 conversion
// ---------------------------------------------------------------------------
__global__ void cvt_f32_f16(const float* __restrict__ in,
                            _Float16* __restrict__ out, int n) {
    int i = blockIdx.x * 256 + threadIdx.x;
    if (i < n) out[i] = (_Float16)in[i];
}

// ---------------------------------------------------------------------------
// QKV GEMM: [8192 x 768] f16  @  W[2304 x 768] f16 (row-major, y = x*W^T) + b
// Wave computes 16(M) x 64(N). Epilogue scatters into q/k/v^T f16 layouts.
//   q : [96][1024][64]   (bh, token, c)
//   k : [96][1024][64]
//   vT: [96][64][1024]   (bh, c, token)   <- transposed for PV B-fragments
// grid: (2304/64=36, 512/4=128), block 128 (4 waves)
// ---------------------------------------------------------------------------
__global__ __launch_bounds__(128)
void qkv_gemm(const _Float16* __restrict__ A, const _Float16* __restrict__ W,
              const float* __restrict__ bias,
              _Float16* __restrict__ qh, _Float16* __restrict__ kh,
              _Float16* __restrict__ vT) {
    int lane = threadIdx.x & 31;
    int wave = threadIdx.x >> 5;
    int n0 = blockIdx.x * 64;
    int m0 = (blockIdx.y * 4 + wave) * 16;
    v8f acc[4] = {};
    for (int k0 = 0; k0 < 768; k0 += 32) {
        v16h a = load_a_tile(A + (size_t)m0 * 768 + k0, 768, lane);
#pragma unroll
        for (int nt = 0; nt < 4; nt++) {
            v16h b = load_b_tile(W + (size_t)(n0 + nt * 16) * 768 + k0, 768, lane);
            acc[nt] = wmma_f16(a, b, acc[nt]);
        }
    }
    int nn = lane & 15, hi = lane >> 4;
#pragma unroll
    for (int nt = 0; nt < 4; nt++) {
        int o = n0 + nt * 16 + nn;
        float bb = bias[o];
        int which = o / 768, rem = o % 768;
        int head = rem >> 6, c = rem & 63;
#pragma unroll
        for (int r = 0; r < 8; r++) {
            int tm = m0 + r + hi * 8;          // token index 0..8191
            int b_ = tm >> 10, hw = tm & 1023;
            int bh = b_ * 12 + head;
            _Float16 h = (_Float16)(acc[nt][r] + bb);
            if (which == 0)      qh[((size_t)bh * 1024 + hw) * 64 + c] = h;
            else if (which == 1) kh[((size_t)bh * 1024 + hw) * 64 + c] = h;
            else                 vT[((size_t)bh * 64 + c) * 1024 + hw] = h;
        }
    }
}

// ---------------------------------------------------------------------------
// Decomposed relative position bias:
//   rel_h[bh][hw][kh] = sum_c q[bh][hw][c] * rel_pos_h[(hw>>5) - kh + 31][c]
//   rel_w[bh][hw][kw] = sum_c q[bh][hw][c] * rel_pos_w[(hw&31) - kw + 31][c]
// grid: 96*1024 blocks of 64 threads; q row staged through LDS.
// ---------------------------------------------------------------------------
__global__ __launch_bounds__(64)
void rel_bias(const _Float16* __restrict__ qh,
              const float* __restrict__ tabh, const float* __restrict__ tabw,
              float* __restrict__ relh, float* __restrict__ relw) {
    __shared__ float qrow[64];
    int bhw = blockIdx.x;
    int bh = bhw >> 10, hw = bhw & 1023;
    int tid = threadIdx.x;
    qrow[tid] = (float)qh[((size_t)bh * 1024 + hw) * 64 + tid];
    __syncthreads();
    if (tid < 32) {
        int idx = (hw >> 5) - tid + 31;
        const float* t = tabh + (size_t)idx * 64;
        float s = 0.f;
#pragma unroll
        for (int c = 0; c < 64; c++) s += qrow[c] * t[c];
        relh[(size_t)bhw * 32 + tid] = s;
    } else {
        int kw = tid - 32;
        int idx = (hw & 31) - kw + 31;
        const float* t = tabw + (size_t)idx * 64;
        float s = 0.f;
#pragma unroll
        for (int c = 0; c < 64; c++) s += qrow[c] * t[c];
        relw[(size_t)bhw * 32 + kw] = s;
    }
}

// ---------------------------------------------------------------------------
// Fused flash attention with decomposed rel-pos bias.
// grid: (96 bh, 16 q-tiles of 64). block 128 = 4 waves; wave owns 16 q rows.
// K/V tiles (32 keys = 4KB each) are staged cooperatively into LDS with
// async global->LDS copies, double-buffered so tile kp+1 streams in while
// tile kp is consumed by WMMA. Per 32-key tile: 4 wmma (QK^T) + online
// softmax + LDS P staging + 4 wmma (PV). Output f16 [token][head*64+c].
// ---------------------------------------------------------------------------
__global__ __launch_bounds__(128)
void attn_flash(const _Float16* __restrict__ qh, const _Float16* __restrict__ kh,
                const _Float16* __restrict__ vT,
                const float* __restrict__ relh, const float* __restrict__ relw,
                _Float16* __restrict__ outh) {
    __shared__ _Float16 kbuf[2][32 * 64];    // [key][c]      4KB per buffer
    __shared__ _Float16 vbuf[2][64 * 32];    // [c][key]      4KB per buffer
    __shared__ _Float16 pst[4][16 * 32];     // per-wave P staging (16x32 f16)
    int tid  = threadIdx.x;
    int lane = tid & 31;
    int wave = tid >> 5;
    int nn = lane & 15, hi = lane >> 4;
    int bh = blockIdx.x;
    int q0 = blockIdx.y * 64 + wave * 16;

    const _Float16* qbase  = qh + ((size_t)bh * 1024 + q0) * 64;
    const _Float16* kgbase = kh + (size_t)bh * 1024 * 64;
    const _Float16* vgbase = vT + (size_t)bh * 64 * 1024;

    v16h qa0 = load_a_tile(qbase + 0,  64, lane);   // c = 0..31
    v16h qa1 = load_a_tile(qbase + 32, 64, lane);   // c = 32..63

    v8f o[4] = {};
    float rm[8], rs[8];
#pragma unroll
    for (int r = 0; r < 8; r++) { rm[r] = -3.0e38f; rs[r] = 0.f; }
    const float scale = 0.125f;   // 64^-0.5

    // Cooperative stage of one 32-key K/V tile into LDS buffer bf.
    // K tile: 2048 contiguous halves. V tile: 64 rows x 32 halves (stride 1024).
    auto stage_kv = [&](int kb, int bf) {
#pragma unroll
        for (int rep = 0; rep < 2; rep++) {
            int j = tid + rep * 128;                    // 0..255, 8 halves each
            cp_g2l_16(kgbase + (size_t)kb * 64 + j * 8, &kbuf[bf][j * 8]);
            int c = j >> 2, seg = j & 3;
            cp_g2l_16(vgbase + (size_t)c * 1024 + kb + seg * 8,
                      &vbuf[bf][c * 32 + seg * 8]);
        }
    };

    stage_kv(0, 0);

    for (int kp = 0; kp < 32; kp++) {        // 32 keys per iteration
        int kb  = kp * 32;
        int cur = kp & 1;
        cp_g2l_wait();                       // this wave's staged copies landed
        __syncthreads();                     // all waves' copies landed; all
                                             // reads of buffer `cur ^ 1` finished
        if (kp + 1 < 32) stage_kv(kb + 32, cur ^ 1);

        v8f s[2];
#pragma unroll
        for (int sub = 0; sub < 2; sub++) {
            const _Float16* kl = &kbuf[cur][(sub * 16 + nn) * 64];
            v16h b0 = *(const v16h*)(kl + hi * 16);        // c = 0..31
            v16h b1 = *(const v16h*)(kl + 32 + hi * 16);   // c = 32..63
            v8f t = {};
            t = wmma_f16(qa0, b0, t);
            t = wmma_f16(qa1, b1, t);
            s[sub] = t;
        }
        // keys kb..kb+31 all map to kh-row == kp; kw = sub*16 + nn
        float tmax[8];
#pragma unroll
        for (int r = 0; r < 8; r++) {
            int m = q0 + r + hi * 8;
            size_t roff = ((size_t)bh * 1024 + m) * 32;
            float bhv = relh[roff + kp];
            float a0 = s[0][r] * scale + bhv + relw[roff + nn];
            float a1 = s[1][r] * scale + bhv + relw[roff + 16 + nn];
            s[0][r] = a0; s[1][r] = a1;
            tmax[r] = fmaxf(a0, a1);
        }
#pragma unroll
        for (int r = 0; r < 8; r++) {
#pragma unroll
            for (int mk = 1; mk < 16; mk <<= 1)
                tmax[r] = fmaxf(tmax[r], __shfl_xor(tmax[r], mk, 32));
        }
        float corr[8], psum[8];
#pragma unroll
        for (int r = 0; r < 8; r++) {
            float nm = fmaxf(rm[r], tmax[r]);
            corr[r] = __expf(rm[r] - nm);
            rm[r] = nm;
            float p0 = __expf(s[0][r] - nm);
            float p1 = __expf(s[1][r] - nm);
            psum[r] = p0 + p1;
            int m = r + hi * 8;
            pst[wave][m * 32 + nn]      = (_Float16)p0;   // key col sub0
            pst[wave][m * 32 + 16 + nn] = (_Float16)p1;   // key col sub1
        }
#pragma unroll
        for (int r = 0; r < 8; r++) {
#pragma unroll
            for (int mk = 1; mk < 16; mk <<= 1)
                psum[r] += __shfl_xor(psum[r], mk, 32);
            rs[r] = rs[r] * corr[r] + psum[r];
        }
#pragma unroll
        for (int nt = 0; nt < 4; nt++)
#pragma unroll
            for (int r = 0; r < 8; r++) o[nt][r] *= corr[r];

        // Reload P in A-fragment layout (same-wave DS ops are in-order).
        {
            const _Float16* pp = &pst[wave][nn * 32 + hi * 16];
            v8h lo = *(const v8h*)pp;
            v8h hh = *(const v8h*)(pp + 8);
            v16h pa = make_a16(lo, hh);
#pragma unroll
            for (int nt = 0; nt < 4; nt++) {
                v16h bv = *(const v16h*)&vbuf[cur][(nt * 16 + nn) * 32 + hi * 16];
                o[nt] = wmma_f16(pa, bv, o[nt]);
            }
        }
    }

    int b_ = bh / 12, head = bh % 12;
#pragma unroll
    for (int nt = 0; nt < 4; nt++)
#pragma unroll
        for (int r = 0; r < 8; r++) {
            int m = q0 + r + hi * 8;
            size_t row = (size_t)b_ * 1024 + m;
            outh[row * 768 + head * 64 + nt * 16 + nn] =
                (_Float16)(o[nt][r] / rs[r]);
        }
}

// ---------------------------------------------------------------------------
// Output projection: [8192 x 768] f16 @ W[768 x 768] f16 + b  -> f32 d_out
// grid: (768/64=12, 512/4=128), block 128
// ---------------------------------------------------------------------------
__global__ __launch_bounds__(128)
void proj_gemm(const _Float16* __restrict__ A, const _Float16* __restrict__ W,
               const float* __restrict__ bias, float* __restrict__ out) {
    int lane = threadIdx.x & 31;
    int wave = threadIdx.x >> 5;
    int n0 = blockIdx.x * 64;
    int m0 = (blockIdx.y * 4 + wave) * 16;
    v8f acc[4] = {};
    for (int k0 = 0; k0 < 768; k0 += 32) {
        v16h a = load_a_tile(A + (size_t)m0 * 768 + k0, 768, lane);
#pragma unroll
        for (int nt = 0; nt < 4; nt++) {
            v16h b = load_b_tile(W + (size_t)(n0 + nt * 16) * 768 + k0, 768, lane);
            acc[nt] = wmma_f16(a, b, acc[nt]);
        }
    }
    int nn = lane & 15, hi = lane >> 4;
#pragma unroll
    for (int nt = 0; nt < 4; nt++) {
        int o = n0 + nt * 16 + nn;
        float bb = bias[o];
#pragma unroll
        for (int r = 0; r < 8; r++) {
            int tm = m0 + r + hi * 8;
            out[(size_t)tm * 768 + o] = acc[nt][r] + bb;
        }
    }
}

// ---------------------------------------------------------------------------
extern "C" void kernel_launch(void* const* d_in, const int* in_sizes, int n_in,
                              void* d_out, int out_size, void* d_ws, size_t ws_size,
                              hipStream_t stream) {
    const float* x      = (const float*)d_in[0];   // (8,32,32,768)
    const float* qkv_w  = (const float*)d_in[1];   // (2304,768)
    const float* qkv_b  = (const float*)d_in[2];   // (2304,)
    const float* proj_w = (const float*)d_in[3];   // (768,768)
    const float* proj_b = (const float*)d_in[4];   // (768,)
    const float* rph    = (const float*)d_in[5];   // (63,64)
    const float* rpw    = (const float*)d_in[6];   // (63,64)

    char* ws = (char*)d_ws;
    size_t off = 0;
    auto carve = [&](size_t bytes) -> void* {
        void* p = ws + off;
        off += (bytes + 255) & ~(size_t)255;
        return p;
    };
    const size_t NTOK = 8192, C = 768, BH = 96, HW = 1024, HD = 64;
    _Float16* x_h    = (_Float16*)carve(NTOK * C * 2);
    _Float16* qkvw_h = (_Float16*)carve(3 * C * C * 2);
    _Float16* projw_h= (_Float16*)carve(C * C * 2);
    _Float16* q_h    = (_Float16*)carve(BH * HW * HD * 2);
    _Float16* k_h    = (_Float16*)carve(BH * HW * HD * 2);
    _Float16* vT_h   = (_Float16*)carve(BH * HW * HD * 2);
    float*    relh   = (float*)   carve(BH * HW * 32 * 4);
    float*    relw   = (float*)   carve(BH * HW * 32 * 4);
    _Float16* ao_h   = (_Float16*)carve(NTOK * C * 2);

    int nx  = (int)(NTOK * C);
    int nw1 = (int)(3 * C * C);
    int nw2 = (int)(C * C);
    cvt_f32_f16<<<(nx  + 255) / 256, 256, 0, stream>>>(x,      x_h,    nx);
    cvt_f32_f16<<<(nw1 + 255) / 256, 256, 0, stream>>>(qkv_w,  qkvw_h, nw1);
    cvt_f32_f16<<<(nw2 + 255) / 256, 256, 0, stream>>>(proj_w, projw_h,nw2);

    qkv_gemm<<<dim3(36, 128), 128, 0, stream>>>(x_h, qkvw_h, qkv_b, q_h, k_h, vT_h);
    rel_bias<<<96 * 1024, 64, 0, stream>>>(q_h, rph, rpw, relh, relw);
    attn_flash<<<dim3(96, 16), 128, 0, stream>>>(q_h, k_h, vT_h, relh, relw, ao_h);
    proj_gemm<<<dim3(12, 128), 128, 0, stream>>>(ao_h, projw_h, proj_b, (float*)d_out);
}
